// DecoderAttention_84404697301288
// MI455X (gfx1250) — matrix operand order repaired and checked by reading, not compiled, not run on hardware
//
#include <hip/hip_runtime.h>
#include <hip/hip_bf16.h>

// ---------------------------------------------------------------------------
// Problem constants (from reference): B=2, S=2048, D=1024, H=16, DK=DV=64
//   M = B*S = 4096 rows for all projection GEMMs, N = K = 1024.
// All matmul work routed through v_wmma_f32_16x16x32_bf16 (wave32 CDNA5).
// A-tile staging: CDNA5 async global->LDS DMA (ASYNCcnt), double-buffered.
// B-tile staging: global->VGPR issued before compute, VGPR->LDS scatter after
// compute, so s_wait_loadcnt lands in the shadow of the WMMAs.
// ---------------------------------------------------------------------------

typedef __attribute__((ext_vector_type(16))) __bf16 v16bf;
typedef __attribute__((ext_vector_type(8)))  __bf16 v8bf;
typedef __attribute__((ext_vector_type(8)))  float  v8f;

union FragB16 { v16bf v; v8bf h[2]; __bf16 e[16]; };
union AccF    { v8f   v; float e[8]; };

static __device__ __forceinline__ v8f wmma_bf16(v16bf a, v16bf b, v8f c) {
  // (neg_a, A, neg_b, B, c_mod, C, reuse_a, reuse_b)
  return __builtin_amdgcn_wmma_f32_16x16x32_bf16(false, a, false, b, (short)0, c,
                                                 false, false);
}

// XOR lane shuffle via ds_swizzle_b32 (group-of-32 mode, and_mask=0x1f).
// Masks 1/2/4/8 stay inside each 16-lane half, matching the WMMA C layout
// where one row's 16 columns live in lanes 0-15 (or 16-31).
#define SWZ_XOR(x, m)                                                          \
  __builtin_bit_cast(float, __builtin_amdgcn_ds_swizzle(                       \
                                __builtin_bit_cast(int, (x)),                  \
                                (0x1f | ((m) << 10))))

static __device__ __forceinline__ float redmax16(float x) {
  x = fmaxf(x, SWZ_XOR(x, 1));
  x = fmaxf(x, SWZ_XOR(x, 2));
  x = fmaxf(x, SWZ_XOR(x, 4));
  x = fmaxf(x, SWZ_XOR(x, 8));
  return x;
}
static __device__ __forceinline__ float redsum16(float x) {
  x += SWZ_XOR(x, 1);
  x += SWZ_XOR(x, 2);
  x += SWZ_XOR(x, 4);
  x += SWZ_XOR(x, 8);
  return x;
}

// ---------------------------------------------------------------------------
// fp32 -> bf16 cast (grid-stride)
// ---------------------------------------------------------------------------
__global__ void cast_f32_bf16(const float* __restrict__ src,
                              __bf16* __restrict__ dst, int n) {
  int i = blockIdx.x * blockDim.x + threadIdx.x;
  int stride = gridDim.x * blockDim.x;
  for (; i < n; i += stride) dst[i] = (__bf16)src[i];
}

// ---------------------------------------------------------------------------
// Tiled bf16 GEMM:  C[M=4096, N=1024] = A[4096,1024] * W[1024,1024] + bias
// 256 threads = 8 waves; block tile 128x128; K-step 32; double-buffered LDS.
// Wave (wm in 0..1, wn in 0..3) owns a 64x32 sub-tile = 4x2 WMMA tiles.
// mode 0: out bf16 laid out [B,H,S,64]   (Q, K)
// mode 1: out bf16 laid out [B,H,64,S]   (V transposed for attention B-frags)
// mode 2: out f32  laid out [M,N]        (final output projection)
// ---------------------------------------------------------------------------
__global__ __launch_bounds__(256) void gemm_bf16(
    const __bf16* __restrict__ A, const __bf16* __restrict__ W,
    const float* __restrict__ bias, void* __restrict__ out, int mode) {
  // padded pitch 40 bf16 (80 B): 16B-aligned fragment chunks, conflict-free
  __shared__ __bf16 Alds[2][128 * 40];   // [buf][m][k]
  __shared__ __bf16 Blds[2][128 * 40];   // [buf][n][k] (transposed)

  const int tid  = threadIdx.x;
  const int lane = tid & 31;
  const int w    = tid >> 5;
  const int wm   = w >> 2;            // 0..1  (64 rows each)
  const int wn   = w & 3;             // 0..3  (32 cols each)
  const int hi   = lane >> 4;         // half-wave select
  const int ln   = lane & 15;
  const int mbase = blockIdx.y * 128;
  const int nbase = blockIdx.x * 128;

  const v8f vzero = {};
  AccF acc[4][2];
#pragma unroll
  for (int mt = 0; mt < 4; ++mt)
#pragma unroll
    for (int nt = 0; nt < 2; ++nt) acc[mt][nt].v = vzero;

  const int arow = tid >> 1, ach = (tid & 1) << 4;   // A: 128 rows x 2 chunks
  const int bk   = tid >> 3, bnc = (tid & 7) << 4;   // W: 32 rows x 8 chunks

  // LDS byte addresses for this thread's A-tile slots (low 32 bits of flat
  // shared-aperture address == LDS offset, ISA 10.2 aperture mapping)
  const unsigned lds_a[2] = {
      (unsigned)(uintptr_t)&Alds[0][arow * 40 + ach],
      (unsigned)(uintptr_t)&Alds[1][arow * 40 + ach]};

  // Issue async global->LDS DMA for one A tile. INST_OFFSET is added to BOTH
  // the LDS and global addresses, so one base pair + offset:16 covers 32 B.
  // No "memory" clobber: writes target the non-current buffer; ordering vs.
  // the publish point is enforced by the volatile s_wait_asynccnt below.
  auto stage_a = [&](int k0, int buf) {
    const __bf16* as = A + (size_t)(mbase + arow) * 1024 + k0 + ach;
    asm volatile(
        "global_load_async_to_lds_b128 %0, %1, off\n\t"
        "global_load_async_to_lds_b128 %0, %1, off offset:16"
        :: "v"(lds_a[buf]), "v"(as));
  };

  // B tile pipelined through registers: load_b issues the global loads,
  // scatter_b (the first USE -> where s_wait_loadcnt goes) runs post-compute.
  FragB16 wv;
  auto load_b = [&](int k0) {
    const __bf16* wsrc = W + (size_t)(k0 + bk) * 1024 + nbase + bnc;
    wv.h[0] = *(const v8bf*)wsrc;
    wv.h[1] = *(const v8bf*)(wsrc + 8);
  };
  auto scatter_b = [&](int buf) {
#pragma unroll
    for (int e = 0; e < 16; ++e) Blds[buf][(bnc + e) * 40 + bk] = wv.e[e];
  };

  // Prologue: stage tile 0, publish.
  load_b(0);
  stage_a(0, 0);
  scatter_b(0);
  asm volatile("s_wait_asynccnt 0x0" ::: "memory");
  __syncthreads();

  for (int k0 = 0; k0 < 1024; k0 += 32) {
    const int cur  = (k0 >> 5) & 1;
    const bool more = (k0 + 32) < 1024;

    // ---- issue tile i+1 traffic (async A DMA + B global loads) ----
    if (more) {
      stage_a(k0 + 32, cur ^ 1);
      load_b(k0 + 32);
      // hint tile i+2 into L2 -> global_prefetch_b8
      __builtin_prefetch(A + (size_t)(mbase + arow) * 1024 + k0 + 64 + ach, 0, 1);
      __builtin_prefetch(W + (size_t)(k0 + 64 + bk) * 1024 + nbase + bnc, 0, 1);
    }

    // ---- compute on tile i (loads above are still in flight) ----
    FragB16 bf[2];
#pragma unroll
    for (int nt = 0; nt < 2; ++nt) {
      const __bf16* bp = &Blds[cur][(wn * 32 + nt * 16 + ln) * 40 + (hi << 3)];
      bf[nt].h[0] = *(const v8bf*)bp;
      bf[nt].h[1] = *(const v8bf*)(bp + 16);
    }
#pragma unroll
    for (int mt = 0; mt < 4; ++mt) {
      FragB16 af;
      const __bf16* ap = &Alds[cur][(wm * 64 + mt * 16 + ln) * 40];
      const int kc = hi << 3;               // lane<16: K{0-7,16-23}; else {8-15,24-31}
      af.h[0] = *(const v8bf*)(ap + kc);
      af.h[1] = *(const v8bf*)(ap + kc + 16);
#pragma unroll
      for (int nt = 0; nt < 2; ++nt)
        acc[mt][nt].v = wmma_bf16(af.v, bf[nt].v, acc[mt][nt].v);
    }

    // ---- B-tile VGPR->LDS scatter (s_wait_loadcnt lands here, post-WMMA) ----
    if (more) scatter_b(cur ^ 1);

    // ---- publish tile i+1: drain async writes, one barrier per iter ----
    asm volatile("s_wait_asynccnt 0x0" ::: "memory");
    __syncthreads();
  }

  // --- epilogue: bias + layout-specific store ---
#pragma unroll
  for (int nt = 0; nt < 2; ++nt) {
    const int n  = nbase + wn * 32 + nt * 16 + ln;
    const float bn = bias[n];
#pragma unroll
    for (int mt = 0; mt < 4; ++mt) {
#pragma unroll
      for (int r = 0; r < 8; ++r) {
        const int m = mbase + wm * 64 + mt * 16 + (hi << 3) + r;
        const float val = acc[mt][nt].e[r] + bn;
        if (mode == 2) {
          ((float*)out)[(size_t)m * 1024 + n] = val;
        } else {
          const int b = m >> 11, s = m & 2047, hh = n >> 6, dk = n & 63;
          size_t idx;
          if (mode == 0)
            idx = (((size_t)(b * 16 + hh) * 2048) + s) * 64 + dk;       // [B,H,S,64]
          else
            idx = (((size_t)(b * 16 + hh) * 64) + dk) * 2048 + s;       // [B,H,64,S]
          ((__bf16*)out)[idx] = (__bf16)val;
        }
      }
    }
  }
}

// ---------------------------------------------------------------------------
// Flash attention: Q[B,H,S,64] x K[B,H,S,64]^T -> softmax -> x Vt[B,H,64,S]
// grid = (S/128, H, B); 8 waves/block, each wave owns 16 query rows.
// Causal: key loop bounded by the wave's max query row.
// ---------------------------------------------------------------------------
__global__ __launch_bounds__(256) void attn_kernel(
    const __bf16* __restrict__ Q, const __bf16* __restrict__ K,
    const __bf16* __restrict__ Vt, const unsigned char* __restrict__ pad,
    __bf16* __restrict__ Oa) {
  __shared__ __bf16 P[8][16 * 40];   // per-wave P-tile staging (pitch 40)

  const int lane = threadIdx.x & 31;
  const int w    = threadIdx.x >> 5;
  const int hi   = lane >> 4;
  const int ln   = lane & 15;
  const int b    = blockIdx.z, h = blockIdx.y;
  const int q0   = blockIdx.x * 128 + w * 16;
  const int kc   = hi << 3;

  // Q fragments (dims 0-31 and 32-63), row = q0 + ln for both lane halves
  const __bf16* qp = Q + (((size_t)(b * 16 + h) * 2048) + q0 + ln) * 64;
  FragB16 a0, a1;
  a0.h[0] = *(const v8bf*)(qp + kc);
  a0.h[1] = *(const v8bf*)(qp + kc + 16);
  a1.h[0] = *(const v8bf*)(qp + 32 + kc);
  a1.h[1] = *(const v8bf*)(qp + 32 + kc + 16);

  const v8f vzero = {};
  AccF o[4];
#pragma unroll
  for (int c = 0; c < 4; ++c) o[c].v = vzero;
  float mrow[8], lrow[8];
#pragma unroll
  for (int r = 0; r < 8; ++r) { mrow[r] = -3.0e38f; lrow[r] = 0.0f; }

  const float scale = 0.125f;   // 1/sqrt(64)
  const int   kmax  = q0 + 15;  // causal bound for this wave's rows
  const __bf16* Kb = K  + ((size_t)(b * 16 + h) * 2048) * 64;
  const __bf16* Vb = Vt + ((size_t)(b * 16 + h) * 64) * 2048;
  const unsigned char* pb = pad + (size_t)b * 2048;

  for (int kb = 0; kb <= kmax; kb += 32) {
    // hint next key block into cache while this one computes
    if (kb + 32 <= kmax) {
      __builtin_prefetch(Kb + (size_t)(kb + 32 + ln) * 64, 0, 1);
      __builtin_prefetch(Vb + (size_t)ln * 2048 + kb + 32, 0, 1);
    }

    // ---- scores: two 16x16 tiles, K-dim 64 = 2 x wmma(K=32) ----
    AccF s[2];
#pragma unroll
    for (int t = 0; t < 2; ++t) {
      const __bf16* kp = Kb + (size_t)(kb + t * 16 + ln) * 64 + (hi << 3);
      FragB16 b0, b1;
      b0.h[0] = *(const v8bf*)kp;        b0.h[1] = *(const v8bf*)(kp + 16);
      b1.h[0] = *(const v8bf*)(kp + 32); b1.h[1] = *(const v8bf*)(kp + 48);
      s[t].v = wmma_bf16(a0.v, b0.v, vzero);
      s[t].v = wmma_bf16(a1.v, b1.v, s[t].v);
    }

    const unsigned char pm0 = pb[kb + ln];
    const unsigned char pm1 = pb[kb + 16 + ln];
    const int k0i = kb + ln, k1i = kb + 16 + ln;

    // ---- online softmax per row (rows r / r+8 by lane half) ----
#pragma unroll
    for (int r = 0; r < 8; ++r) {
      const int row = q0 + (hi << 3) + r;
      float v0 = s[0].e[r] * scale;
      float v1 = s[1].e[r] * scale;
      if (k0i > row || pm0) v0 = -3.0e38f;
      if (k1i > row || pm1) v1 = -3.0e38f;
      const float rm = redmax16(fmaxf(v0, v1));
      const float mn = fmaxf(mrow[r], rm);
      const float al = __expf(mrow[r] - mn);
      const float p0 = __expf(v0 - mn);
      const float p1 = __expf(v1 - mn);
      lrow[r] = lrow[r] * al + redsum16(p0 + p1);
      mrow[r] = mn;
#pragma unroll
      for (int c = 0; c < 4; ++c) o[c].e[r] *= al;
      P[w][(r + (hi << 3)) * 40 + ln]      = (__bf16)p0;
      P[w][(r + (hi << 3)) * 40 + 16 + ln] = (__bf16)p1;
    }
    // per-wave LDS: DS ops are in-order, but fence the compiler + counter
    asm volatile("s_wait_dscnt 0" ::: "memory");

    // ---- reload P as A-fragment (16 rows x 32 keys) ----
    FragB16 pa;
    const __bf16* pp = &P[w][ln * 40 + kc];
    pa.h[0] = *(const v8bf*)pp;
    pa.h[1] = *(const v8bf*)(pp + 16);

    // ---- O += P x V  (4 dv-chunks of 16) ----
#pragma unroll
    for (int c = 0; c < 4; ++c) {
      const __bf16* vp = Vb + (size_t)(c * 16 + ln) * 2048 + kb + (hi << 3);
      FragB16 vf;
      vf.h[0] = *(const v8bf*)vp;
      vf.h[1] = *(const v8bf*)(vp + 16);
      o[c].v = wmma_bf16(pa.v, vf.v, o[c].v);
    }
  }

  // ---- normalize and write Oa[b, s, h*64+dv] (bf16) ----
#pragma unroll
  for (int c = 0; c < 4; ++c) {
    const int dv = c * 16 + ln;
#pragma unroll
    for (int r = 0; r < 8; ++r) {
      const int row = q0 + (hi << 3) + r;
      const float val = o[c].e[r] / fmaxf(lrow[r], 1e-20f);
      Oa[((size_t)b * 2048 + row) * 1024 + h * 64 + dv] = (__bf16)val;
    }
  }
}

// ---------------------------------------------------------------------------
// Host launch
// ---------------------------------------------------------------------------
extern "C" void kernel_launch(void* const* d_in, const int* in_sizes, int n_in,
                              void* d_out, int out_size, void* d_ws,
                              size_t ws_size, hipStream_t stream) {
  const float* x  = (const float*)d_in[0];
  const unsigned char* mask = (const unsigned char*)d_in[1];
  const float* Wq = (const float*)d_in[2];
  const float* bq = (const float*)d_in[3];
  const float* Wk = (const float*)d_in[4];
  const float* bk = (const float*)d_in[5];
  const float* Wv = (const float*)d_in[6];
  const float* bv = (const float*)d_in[7];
  const float* Wo = (const float*)d_in[8];
  const float* bo = (const float*)d_in[9];

  char* ws = (char*)d_ws;
  size_t off = 0;
  auto alloc = [&](size_t bytes) -> void* {
    void* p = ws + off;
    off += (bytes + 255) & ~(size_t)255;
    return p;
  };
  const size_t MK = 4096ull * 1024;       // x rows * D
  const size_t WK = 1024ull * 1024;       // weight elems
  const size_t QK = 2ull * 16 * 2048 * 64;

  __bf16* xb  = (__bf16*)alloc(MK * 2);
  __bf16* Wqb = (__bf16*)alloc(WK * 2);
  __bf16* Wkb = (__bf16*)alloc(WK * 2);
  __bf16* Wvb = (__bf16*)alloc(WK * 2);
  __bf16* Wob = (__bf16*)alloc(WK * 2);
  __bf16* Qb  = (__bf16*)alloc(QK * 2);
  __bf16* Kb  = (__bf16*)alloc(QK * 2);
  __bf16* Vtb = (__bf16*)alloc(QK * 2);
  __bf16* Oab = (__bf16*)alloc(MK * 2);

  cast_f32_bf16<<<1024, 256, 0, stream>>>(x, xb, (int)MK);
  cast_f32_bf16<<<256, 256, 0, stream>>>(Wq, Wqb, (int)WK);
  cast_f32_bf16<<<256, 256, 0, stream>>>(Wk, Wkb, (int)WK);
  cast_f32_bf16<<<256, 256, 0, stream>>>(Wv, Wvb, (int)WK);
  cast_f32_bf16<<<256, 256, 0, stream>>>(Wo, Wob, (int)WK);

  dim3 gg(8, 32);   // N/128, M/128
  gemm_bf16<<<gg, 256, 0, stream>>>(xb, Wqb, bq, Qb, 0);
  gemm_bf16<<<gg, 256, 0, stream>>>(xb, Wkb, bk, Kb, 0);
  gemm_bf16<<<gg, 256, 0, stream>>>(xb, Wvb, bv, Vtb, 1);

  dim3 ga(16, 16, 2);  // q-blocks, H, B
  attn_kernel<<<ga, 256, 0, stream>>>(Qb, Kb, Vtb, mask, Oab);

  gemm_bf16<<<gg, 256, 0, stream>>>(Oab, Wob, bo, d_out, 2);
}
